// EarlyDenseCellDiffPool_81655918231967
// MI455X (gfx1250) — matrix operand order repaired and checked by reading, not compiled, not run on hardware
//
#include <hip/hip_runtime.h>
#include <hip/hip_bf16.h>
#include <math.h>

// ---------------------------------------------------------------------------
// EarlyDenseCellDiffPool for MI455X (gfx1250).
//  * All GEMMs on v_wmma_f32_16x16x32_bf16 (fp32 accumulate), 64x64x32 block
//    tiles, 4 waves/block, bf16 tiles staged in LDS, fragments loaded as
//    ds_load_b128 pairs.
//  * Interior tiles use global_load_async_to_lds_b32 + s_wait_asynccnt
//    (ASYNCcnt pipeline) when the toolchain exposes the builtins.
//  * ||A - S S^T||_F fused as a WMMA tile kernel (S S^T never materialized).
// ---------------------------------------------------------------------------

typedef __attribute__((ext_vector_type(16))) __bf16 v16bf;
typedef __attribute__((ext_vector_type(8)))  __bf16 v8bf;
typedef __attribute__((ext_vector_type(8)))  float  v8f;

typedef __attribute__((address_space(1))) int as1_int;   // global
typedef __attribute__((address_space(3))) int as3_int;   // LDS

#define WAVE 32
#define BM 64
#define BN 64
#define BK 32
#define NTHREADS 128

#if defined(__has_builtin)
#if __has_builtin(__builtin_amdgcn_global_load_async_to_lds_b32) && \
    __has_builtin(__builtin_amdgcn_s_wait_asynccnt)
#define HAVE_ASYNC_LDS 1
#endif
#endif
#ifndef HAVE_ASYNC_LDS
#define HAVE_ASYNC_LDS 0
#endif

// ------------------------- generic WMMA GEMM -------------------------------
// C[b][m][n] = sum_k A[b][m][k] * B[b][k][n] (+ bias[n]); A optionally stored
// transposed (A[b][k][m]), used for S^T X and S^T A.
template <int TRANSA>
__global__ __launch_bounds__(NTHREADS)
void gemm_wmma_kernel(const float* __restrict__ A, const float* __restrict__ Bm,
                      const float* __restrict__ bias, float* __restrict__ C,
                      int M, int Nc, int K,
                      long long sA, long long sB, long long sC,
                      int lda, int ldb, int ldc) {
#if HAVE_ASYNC_LDS
  __shared__ float sAf[BM * BK];   // raw fp32 staging (async path)
  __shared__ float sBf[BK * BN];
#endif
  __shared__ __align__(16) __bf16 sAb[BM][BK];  // A tile, [m][k]
  __shared__ __align__(16) __bf16 sBb[BN][BK];  // B tile transposed, [n][k]

  const int tid = threadIdx.x;
  const int lane = tid & 31;
  const int w = tid >> 5;
  const int i0 = blockIdx.x * BM, j0 = blockIdx.y * BN, b = blockIdx.z;
  const float* Ab = A + (long long)b * sA;
  const float* Bb = Bm + (long long)b * sB;
  float* Cb = C + (long long)b * sC;

  v8f acc[4];
#pragma unroll
  for (int j = 0; j < 4; ++j) acc[j] = (v8f){0.f, 0.f, 0.f, 0.f, 0.f, 0.f, 0.f, 0.f};

  const int mrow = lane & 15;
  const int kh = lane >> 4;

  for (int k0 = 0; k0 < K; k0 += BK) {
#if HAVE_ASYNC_LDS
    const bool fullA = (i0 + BM <= M) && (k0 + BK <= K);
    const bool fullB = (j0 + BN <= Nc) && (k0 + BK <= K);
    if (fullA) {
      // async global->LDS, coalesced along the memory-contiguous dimension
      for (int t = tid; t < BM * BK; t += NTHREADS) {
        long long g = TRANSA ? ((long long)(k0 + (t >> 6)) * lda + i0 + (t & 63))
                             : ((long long)(i0 + (t >> 5)) * lda + k0 + (t & 31));
        __builtin_amdgcn_global_load_async_to_lds_b32(
            (as1_int*)(Ab + g), (as3_int*)&sAf[t], 0, 0);
      }
    }
    if (fullB) {
      for (int t = tid; t < BK * BN; t += NTHREADS) {
        long long g = (long long)(k0 + (t >> 6)) * ldb + j0 + (t & 63);
        __builtin_amdgcn_global_load_async_to_lds_b32(
            (as1_int*)(Bb + g), (as3_int*)&sBf[t], 0, 0);
      }
    }
    const bool syncA = !fullA, syncB = !fullB;
#else
    const bool syncA = true, syncB = true;
#endif
    if (syncA) {  // edge tiles: bounds-checked load + convert + store
      for (int t = tid; t < BM * BK; t += NTHREADS) {
        int m = TRANSA ? (t & 63) : (t >> 5);
        int kk = TRANSA ? (t >> 6) : (t & 31);
        float v = 0.f;
        if (i0 + m < M && k0 + kk < K)
          v = TRANSA ? Ab[(long long)(k0 + kk) * lda + i0 + m]
                     : Ab[(long long)(i0 + m) * lda + k0 + kk];
        sAb[m][kk] = (__bf16)v;
      }
    }
    if (syncB) {
      for (int t = tid; t < BK * BN; t += NTHREADS) {
        int nn = t & 63, kk = t >> 6;
        float v = 0.f;
        if (j0 + nn < Nc && k0 + kk < K)
          v = Bb[(long long)(k0 + kk) * ldb + j0 + nn];
        sBb[nn][kk] = (__bf16)v;
      }
    }
#if HAVE_ASYNC_LDS
    if (fullA || fullB) __builtin_amdgcn_s_wait_asynccnt(0);
    if (fullA) {  // each thread converts exactly the elements it async-loaded
      for (int t = tid; t < BM * BK; t += NTHREADS) {
        int m = TRANSA ? (t & 63) : (t >> 5);
        int kk = TRANSA ? (t >> 6) : (t & 31);
        sAb[m][kk] = (__bf16)sAf[t];
      }
    }
    if (fullB) {
      for (int t = tid; t < BK * BN; t += NTHREADS)
        sBb[t & 63][t >> 6] = (__bf16)sBf[t];
    }
#endif
    __syncthreads();

    // fragments: two contiguous 16B LDS reads each (ds_load_b128)
    v8bf alo = *(const v8bf*)&sAb[w * 16 + mrow][kh * 8];
    v8bf ahi = *(const v8bf*)&sAb[w * 16 + mrow][16 + kh * 8];
    v16bf af = __builtin_shufflevector(alo, ahi, 0, 1, 2, 3, 4, 5, 6, 7,
                                       8, 9, 10, 11, 12, 13, 14, 15);
#pragma unroll
    for (int j = 0; j < 4; ++j) {
      v8bf blo = *(const v8bf*)&sBb[j * 16 + mrow][kh * 8];
      v8bf bhi = *(const v8bf*)&sBb[j * 16 + mrow][16 + kh * 8];
      v16bf bfv = __builtin_shufflevector(blo, bhi, 0, 1, 2, 3, 4, 5, 6, 7,
                                          8, 9, 10, 11, 12, 13, 14, 15);
      acc[j] = __builtin_amdgcn_wmma_f32_16x16x32_bf16(false, af, false, bfv,
                                                       (short)0, acc[j], false, false);
    }
    __syncthreads();
  }

  // epilogue: wave w owns rows [i0+16w, i0+16w+16)
#pragma unroll
  for (int j = 0; j < 4; ++j) {
    const int n = j0 + j * 16 + mrow;
#pragma unroll
    for (int r = 0; r < 8; ++r) {
      int m = i0 + w * 16 + r + (kh << 3);
      if (m < M && n < Nc) {
        float v = acc[j][r];
        if (bias) v += bias[n];
        Cb[(long long)m * ldc + n] = v;
      }
    }
  }
}

// ---------------- fused link loss: sum((adj - S S^T)^2) --------------------
__global__ __launch_bounds__(WAVE)
void link_loss_kernel(const float* __restrict__ s, const float* __restrict__ adj,
                      int Nn, int Kc, long long sBS, long long adjBS,
                      float* __restrict__ acc) {
  __shared__ __align__(16) __bf16 Al[16][32];
  __shared__ __align__(16) __bf16 Bl[16][32];   // S^T tile, [n][k]
  __shared__ float red[WAVE];
  const int lane = threadIdx.x;
  const int i0 = blockIdx.x * 16, j0 = blockIdx.y * 16, b = blockIdx.z;
  const float* sb = s + (long long)b * sBS;
  const float* ab = adj + (long long)b * adjBS;

  v8f accv = {0.f, 0.f, 0.f, 0.f, 0.f, 0.f, 0.f, 0.f};
  const int mrow = lane & 15;
  const int kh = lane >> 4;

  for (int k0 = 0; k0 < Kc; k0 += 32) {
    for (int t = lane; t < 512; t += WAVE) {
      int r = t >> 5, kk = t & 31;
      float va = (i0 + r < Nn && k0 + kk < Kc)
                     ? sb[(long long)(i0 + r) * Kc + k0 + kk] : 0.f;
      Al[r][kk] = (__bf16)va;
      float vb = (j0 + r < Nn && k0 + kk < Kc)
                     ? sb[(long long)(j0 + r) * Kc + k0 + kk] : 0.f;
      Bl[r][kk] = (__bf16)vb;
    }
    __syncthreads();
    v8bf alo = *(const v8bf*)&Al[mrow][kh * 8];
    v8bf ahi = *(const v8bf*)&Al[mrow][16 + kh * 8];
    v16bf af = __builtin_shufflevector(alo, ahi, 0, 1, 2, 3, 4, 5, 6, 7,
                                       8, 9, 10, 11, 12, 13, 14, 15);
    v8bf blo = *(const v8bf*)&Bl[mrow][kh * 8];
    v8bf bhi = *(const v8bf*)&Bl[mrow][16 + kh * 8];
    v16bf bfv = __builtin_shufflevector(blo, bhi, 0, 1, 2, 3, 4, 5, 6, 7,
                                        8, 9, 10, 11, 12, 13, 14, 15);
    accv = __builtin_amdgcn_wmma_f32_16x16x32_bf16(false, af, false, bfv,
                                                   (short)0, accv, false, false);
    __syncthreads();
  }

  float part = 0.f;
  const int n = j0 + mrow;
#pragma unroll
  for (int r = 0; r < 8; ++r) {
    int m = i0 + r + (kh << 3);
    if (m < Nn && n < Nn) {
      float d = ab[(long long)m * Nn + n] - accv[r];
      part += d * d;
    }
  }
  red[lane] = part;
  __syncthreads();
  if (lane == 0) {
    float t = 0.f;
    for (int k = 0; k < WAVE; ++k) t += red[k];
    atomicAdd(acc, t);
  }
}

// ---------------------- attention scores + softmax -------------------------
__global__ void att_softmax_kernel(const float* __restrict__ adj,
                                   const float* __restrict__ asrc,
                                   const float* __restrict__ adst,
                                   float* __restrict__ att, int Nn,
                                   long long adjBS, long long attBS) {
  __shared__ float sh[256];
  const int i = blockIdx.x, b = blockIdx.y, tid = threadIdx.x;
  const float* arow = adj + (long long)b * adjBS + (long long)i * Nn;
  float* orow = att + (long long)b * attBS + (long long)i * Nn;
  const float* as = asrc + (long long)b * Nn;
  const float ad = adst[(long long)b * Nn + i];

  float mx = -3.4e38f;
  for (int j = tid; j < Nn; j += blockDim.x) {
    if (arow[j] != 0.0f || j == i) {            // adj mask | add_loop diagonal
      float v = ad + as[j];
      v = v >= 0.f ? v : 0.2f * v;              // leaky relu
      mx = fmaxf(mx, v);
    }
  }
  sh[tid] = mx; __syncthreads();
  for (int s = 128; s > 0; s >>= 1) { if (tid < s) sh[tid] = fmaxf(sh[tid], sh[tid + s]); __syncthreads(); }
  mx = sh[0]; __syncthreads();

  float sm = 0.f;
  for (int j = tid; j < Nn; j += blockDim.x) {
    if (arow[j] != 0.0f || j == i) {
      float v = ad + as[j]; v = v >= 0.f ? v : 0.2f * v;
      sm += __expf(v - mx);
    }
  }
  sh[tid] = sm; __syncthreads();
  for (int s = 128; s > 0; s >>= 1) { if (tid < s) sh[tid] += sh[tid + s]; __syncthreads(); }
  sm = sh[0];
  const float inv = 1.f / sm;
  for (int j = tid; j < Nn; j += blockDim.x) {
    float o = 0.f;
    if (arow[j] != 0.0f || j == i) {
      float v = ad + as[j]; v = v >= 0.f ? v : 0.2f * v;
      o = __expf(v - mx) * inv;
    }
    orow[j] = o;
  }
}

// a_src[r] = h[r,:].att_src ; a_dst[r] = h[r,:].att_dst
__global__ void rowdot_kernel(const float* __restrict__ h,
                              const float* __restrict__ wsrc,
                              const float* __restrict__ wdst,
                              float* __restrict__ asrc, float* __restrict__ adst,
                              int C, long long rows) {
  long long r = (long long)blockIdx.x * blockDim.x + threadIdx.x;
  if (r >= rows) return;
  const float* row = h + r * C;
  float s = 0.f, d = 0.f;
  for (int c = 0; c < C; ++c) { float v = row[c]; s += v * wsrc[c]; d += v * wdst[c]; }
  asrc[r] = s; adst[r] = d;
}

// BatchNorm1d (training stats over all rows) + ReLU, in place; 1 block/channel
__global__ void bn_relu_kernel(float* __restrict__ x, const float* __restrict__ gamma,
                               const float* __restrict__ beta, int C, long long rows) {
  __shared__ float sh[256];
  __shared__ float sh2[256];
  const int c = blockIdx.x, tid = threadIdx.x;
  float s = 0.f, ss = 0.f;
  for (long long r = tid; r < rows; r += blockDim.x) {
    float v = x[r * C + c]; s += v; ss += v * v;
  }
  sh[tid] = s; sh2[tid] = ss; __syncthreads();
  for (int k = 128; k > 0; k >>= 1) {
    if (tid < k) { sh[tid] += sh[tid + k]; sh2[tid] += sh2[tid + k]; }
    __syncthreads();
  }
  const float mean = sh[0] / (float)rows;
  const float var = sh2[0] / (float)rows - mean * mean;   // biased, as torch
  const float scale = gamma[c] * rsqrtf(var + 1e-5f);
  const float shift = beta[c] - mean * scale;
  for (long long r = tid; r < rows; r += blockDim.x) {
    float v = x[r * C + c] * scale + shift;
    x[r * C + c] = v > 0.f ? v : 0.f;
  }
}

__global__ void avg2_kernel(const float* __restrict__ a, const float* __restrict__ b,
                            float* __restrict__ o, long long n) {
  long long i = (long long)blockIdx.x * blockDim.x + threadIdx.x;
  long long st = (long long)gridDim.x * blockDim.x;
  for (; i < n; i += st) o[i] = 0.5f * (a[i] + b[i]);
}

// row softmax over Kc (in place) + entropy accumulation
__global__ void s_softmax_ent_kernel(float* __restrict__ s, int Kc,
                                     float* __restrict__ entAcc) {
  __shared__ float sh[128];
  const long long row = blockIdx.x;
  const int tid = threadIdx.x;
  float* p = s + row * (long long)Kc;
  float mx = -3.4e38f;
  for (int k = tid; k < Kc; k += blockDim.x) mx = fmaxf(mx, p[k]);
  sh[tid] = mx; __syncthreads();
  for (int k = 64; k > 0; k >>= 1) { if (tid < k) sh[tid] = fmaxf(sh[tid], sh[tid + k]); __syncthreads(); }
  mx = sh[0]; __syncthreads();
  float sm = 0.f;
  for (int k = tid; k < Kc; k += blockDim.x) sm += __expf(p[k] - mx);
  sh[tid] = sm; __syncthreads();
  for (int k = 64; k > 0; k >>= 1) { if (tid < k) sh[tid] += sh[tid + k]; __syncthreads(); }
  sm = sh[0]; __syncthreads();
  const float inv = 1.f / sm;
  float ent = 0.f;
  for (int k = tid; k < Kc; k += blockDim.x) {
    float q = __expf(p[k] - mx) * inv;
    p[k] = q;
    ent += -q * __logf(q + 1e-15f);
  }
  sh[tid] = ent; __syncthreads();
  for (int k = 64; k > 0; k >>= 1) { if (tid < k) sh[tid] += sh[tid + k]; __syncthreads(); }
  if (tid == 0) atomicAdd(entAcc, sh[0]);
}

__global__ void colmean_kernel(const float* __restrict__ z, float* __restrict__ out,
                               int Nn, int C, int Bn) {
  int t = blockIdx.x * blockDim.x + threadIdx.x;
  if (t >= Bn * C) return;
  int b = t / C, c = t % C;
  const float* base = z + (long long)b * Nn * C + c;
  float s = 0.f;
  for (int i = 0; i < Nn; ++i) s += base[(long long)i * C];
  out[t] = s / (float)Nn;
}

// tiny dense head: out[b,t] = in[b,:] @ w[:,t] + bias[t]
__global__ void head_kernel(const float* __restrict__ in, const float* __restrict__ w,
                            const float* __restrict__ bias, float* __restrict__ out,
                            int C, int T, int Bn) {
  int t = blockIdx.x * blockDim.x + threadIdx.x;
  if (t >= Bn * T) return;
  int b = t / T, tt = t % T;
  float s = bias[tt];
  for (int c = 0; c < C; ++c) s += in[b * C + c] * w[c * T + tt];
  out[t] = s;
}

__global__ void init_acc_kernel(float* acc) {
  if (threadIdx.x < 4) acc[threadIdx.x] = 0.f;
}

__global__ void finalize_kernel(const float* __restrict__ acc, float* __restrict__ out,
                                int N0, int K0n, int Bn) {
  out[4] = sqrtf(acc[0]) / ((float)Bn * (float)N0 * (float)N0);   // link 0
  out[5] = sqrtf(acc[1]) / ((float)Bn * (float)K0n * (float)K0n); // link 1
  out[6] = acc[2] / ((float)Bn * (float)N0);                      // ent 0
  out[7] = acc[3] / ((float)Bn * (float)K0n);                     // ent 1
}

// ---------------------------------------------------------------------------
struct GatP {
  const float *att_dst, *att_src, *beta, *bias, *gamma, *w;
};
static inline GatP gp(void* const* d, int base) {
  GatP p;
  p.att_dst = (const float*)d[base + 0];
  p.att_src = (const float*)d[base + 1];
  p.beta    = (const float*)d[base + 2];
  p.bias    = (const float*)d[base + 3];
  p.gamma   = (const float*)d[base + 4];
  p.w       = (const float*)d[base + 5];
  return p;
}

static inline void launch_gemm(hipStream_t st, const float* A, const float* Bm,
                               const float* bias, float* C, int M, int Nc, int K,
                               long long sA, long long sB, long long sC,
                               int lda, int ldb, int ldc, int batches, bool transA) {
  dim3 grid((M + BM - 1) / BM, (Nc + BN - 1) / BN, batches);
  if (transA)
    gemm_wmma_kernel<1><<<grid, NTHREADS, 0, st>>>(A, Bm, bias, C, M, Nc, K, sA, sB, sC, lda, ldb, ldc);
  else
    gemm_wmma_kernel<0><<<grid, NTHREADS, 0, st>>>(A, Bm, bias, C, M, Nc, K, sA, sB, sC, lda, ldb, ldc);
}

extern "C" void kernel_launch(void* const* d_in, const int* in_sizes, int n_in,
                              void* d_out, int out_size, void* d_ws, size_t ws_size,
                              hipStream_t stream) {
  (void)in_sizes; (void)n_in; (void)out_size; (void)ws_size;
  const int Bn = 2, N = 2048, CIN = 64, H = 128, K0 = 204, TD = 2;
  const long long NN = (long long)N * N;
  const long long BNN = (long long)Bn * NN;
  const long long BNH = (long long)Bn * N * H;
  const long long BNK = (long long)Bn * N * K0;
  const long long BKN = (long long)Bn * K0 * N;
  const long long BKH = (long long)Bn * K0 * H;
  const long long KK = (long long)K0 * K0;
  const long long BKK = (long long)Bn * KK;

  // params flattened after {x, adj_ppi, adj_reg, mask} in sorted-key pytree order
  const int P_EMBED_0_0 = 4, P_EMBED_0_1 = 10, P_EMBED_1_0 = 16, P_EMBED_1_1 = 22;
  const int P_GAT_PPI_0 = 28, P_GAT_PPI_1 = 34, P_GAT_REG_0 = 40, P_GAT_REG_1 = 46;
  const int P_HEAD0_B = 52, P_HEAD0_W = 53, P_HEAD1_B = 54, P_HEAD1_W = 55;
  const int P_LIN_B = 56, P_LIN_W = 57;
  const int P_POOL_0_0 = 58, P_POOL_0_1 = 64, P_POOL_1_0 = 70, P_POOL_1_1 = 76;

  const float* x = (const float*)d_in[0];
  const float* adj_ppi = (const float*)d_in[1];
  const float* adj_reg = (const float*)d_in[2];
  float* out = (float*)d_out;

  // workspace layout (floats)
  float* f = (float*)d_ws;
  float* ATT = f;              // BNN   attention scratch (also level-1 att)
  float* CADJ = ATT + BNN;     // BNN   combined adjacency, level 0
  float* HB = CADJ + BNN;      // BNH   h = x@W scratch (C_out <= 128)
  float* T0 = HB + BNH;        // BNH
  float* T1 = T0 + BNH;        // BNH
  float* Zb = T1 + BNH;        // BNH
  float* XC = Zb + BNH;        // BNH   combined embedding
  float* HBK = XC + BNH;       // BNK   h scratch for C_out = 204
  float* SB = HBK + BNK;       // BNK   assignment S, level 0
  float* T2 = SB + BNK;        // BKN   S^T A
  float* X1 = T2 + BKN;        // BKH   pooled x, level 1
  float* ADJ1 = X1 + BKH;      // BKK   pooled adjacency
  float* Z1 = ADJ1 + BKK;      // BKH
  float* S1 = Z1 + BKH;        // Bn*K0 assignment S, level 1 (Kc = 1)
  float* X2 = S1 + (long long)Bn * K0;  // Bn*H
  float* CX = X2 + (long long)Bn * H;   // Bn*H
  float* ASRC = CX + (long long)Bn * H; // Bn*N
  float* ADST = ASRC + (long long)Bn * N;
  float* ACC = ADST + (long long)Bn * N; // 4: link0, link1, ent0, ent1

  init_acc_kernel<<<1, 32, 0, stream>>>(ACC);

  // one DenseGATConv + BN + ReLU
  auto gat_layer = [&](const float* xin, int Cin, int Cout, const float* adj,
                       long long adjBS, int Nn, const GatP& p, float* hbuf,
                       float* attbuf, long long attBS, float* xout) {
    long long rows = (long long)Bn * Nn;
    launch_gemm(stream, xin, p.w, nullptr, hbuf, Nn, Cout, Cin,
                (long long)Nn * Cin, 0, (long long)Nn * Cout, Cin, Cout, Cout, Bn, false);
    rowdot_kernel<<<(int)((rows + 255) / 256), 256, 0, stream>>>(
        hbuf, p.att_src, p.att_dst, ASRC, ADST, Cout, rows);
    att_softmax_kernel<<<dim3(Nn, Bn), 256, 0, stream>>>(
        adj, ASRC, ADST, attbuf, Nn, adjBS, attBS);
    launch_gemm(stream, attbuf, hbuf, p.bias, xout, Nn, Cout, Nn,
                attBS, (long long)Nn * Cout, (long long)Nn * Cout, Nn, Cout, Cout, Bn, false);
    bn_relu_kernel<<<Cout, 256, 0, stream>>>(xout, p.gamma, p.beta, Cout, rows);
  };

  // ---- per-graph GAT stacks ----
  gat_layer(x, CIN, H, adj_ppi, NN, N, gp(d_in, P_GAT_PPI_0), HB, ATT, NN, T0);
  gat_layer(T0, H, H, adj_ppi, NN, N, gp(d_in, P_GAT_PPI_1), HB, ATT, NN, T1);
  gat_layer(x, CIN, H, adj_reg, NN, N, gp(d_in, P_GAT_REG_0), HB, ATT, NN, T0);
  gat_layer(T0, H, H, adj_reg, NN, N, gp(d_in, P_GAT_REG_1), HB, ATT, NN, Zb);
  avg2_kernel<<<4096, 256, 0, stream>>>(T1, Zb, XC, BNH);
  avg2_kernel<<<32768, 256, 0, stream>>>(adj_ppi, adj_reg, CADJ, BNN);

  // ---- level 0: pool / embed stacks ----
  gat_layer(XC, H, H, CADJ, NN, N, gp(d_in, P_POOL_0_0), HB, ATT, NN, T0);
  gat_layer(T0, H, K0, CADJ, NN, N, gp(d_in, P_POOL_0_1), HBK, ATT, NN, SB);
  gat_layer(XC, H, H, CADJ, NN, N, gp(d_in, P_EMBED_0_0), HB, ATT, NN, T0);
  gat_layer(T0, H, H, CADJ, NN, N, gp(d_in, P_EMBED_0_1), HB, ATT, NN, Zb);

  colmean_kernel<<<(Bn * H + 255) / 256, 256, 0, stream>>>(Zb, CX, N, H, Bn);
  head_kernel<<<1, 64, 0, stream>>>(CX, (const float*)d_in[P_HEAD0_W],
                                    (const float*)d_in[P_HEAD0_B], out + 8, H, TD, Bn);

  // ---- diffpool level 0 ----
  s_softmax_ent_kernel<<<Bn * N, 128, 0, stream>>>(SB, K0, ACC + 2);
  launch_gemm(stream, SB, Zb, nullptr, X1, K0, H, N,
              (long long)N * K0, (long long)N * H, (long long)K0 * H, K0, H, H, Bn, true);
  launch_gemm(stream, SB, CADJ, nullptr, T2, K0, N, N,
              (long long)N * K0, NN, (long long)K0 * N, K0, N, N, Bn, true);
  launch_gemm(stream, T2, SB, nullptr, ADJ1, K0, K0, N,
              (long long)K0 * N, (long long)N * K0, KK, N, K0, K0, Bn, false);
  link_loss_kernel<<<dim3(N / 16, N / 16, Bn), WAVE, 0, stream>>>(
      SB, CADJ, N, K0, (long long)N * K0, NN, ACC + 0);

  // ---- level 1 (204 nodes): pool / embed stacks ----
  gat_layer(X1, H, H, ADJ1, KK, K0, gp(d_in, P_POOL_1_0), HB, ATT, KK, T0);
  gat_layer(T0, H, 1, ADJ1, KK, K0, gp(d_in, P_POOL_1_1), HB, ATT, KK, S1);
  gat_layer(X1, H, H, ADJ1, KK, K0, gp(d_in, P_EMBED_1_0), HB, ATT, KK, T0);
  gat_layer(T0, H, H, ADJ1, KK, K0, gp(d_in, P_EMBED_1_1), HB, ATT, KK, Z1);

  colmean_kernel<<<(Bn * H + 255) / 256, 256, 0, stream>>>(Z1, CX, K0, H, Bn);
  head_kernel<<<1, 64, 0, stream>>>(CX, (const float*)d_in[P_HEAD1_W],
                                    (const float*)d_in[P_HEAD1_B], out + 12, H, TD, Bn);

  // ---- diffpool level 1 (K_c = 1; pooled adjacency unused downstream) ----
  s_softmax_ent_kernel<<<Bn * K0, 128, 0, stream>>>(S1, 1, ACC + 3);
  launch_gemm(stream, S1, Z1, nullptr, X2, 1, H, K0,
              (long long)K0, (long long)K0 * H, (long long)H, 1, H, H, Bn, true);
  link_loss_kernel<<<dim3((K0 + 15) / 16, (K0 + 15) / 16, Bn), WAVE, 0, stream>>>(
      S1, ADJ1, K0, 1, (long long)K0, KK, ACC + 1);

  // ---- final linear head + loss packing ----
  head_kernel<<<1, 64, 0, stream>>>(X2, (const float*)d_in[P_LIN_W],
                                    (const float*)d_in[P_LIN_B], out + 0, H, TD, Bn);
  finalize_kernel<<<1, 1, 0, stream>>>(ACC, out, N, K0, Bn);
}
